// MLA_3925600108584
// MI455X (gfx1250) — matrix operand order repaired and checked by reading, compiled, not run on hardware
//
#include <hip/hip_runtime.h>
#include <stdint.h>

#define DEV_INLINE __device__ __forceinline__

typedef __attribute__((ext_vector_type(16))) __bf16 v16bf;
typedef __attribute__((ext_vector_type(4)))  __bf16 bf16x4;
typedef __attribute__((ext_vector_type(8)))  float  v8f;

constexpr int BB = 2, SS = 2048, HH = 16, DHH = 128, DVV = 128, RR = 512;
#define SCALE_F 0.08838834764831845f /* 128^-0.5 */
#define SCALE_LOG2 (0.08838834764831845f * 1.4426950408889634f) /* into exp2 domain */

// Padded LDS row strides (bf16 elements). Byte strides 80 / 272 are 16B-aligned
// and sweep distinct 4-bank groups across rows -> conflict-free b128 access.
constexpr int LDA = 40;   // 32-elem rows padded to 80B
constexpr int LDK = 136;  // 128-elem rows padded to 272B

DEV_INLINE v8f wmma_bf16(v16bf a, v16bf b, v8f c) {
  return __builtin_amdgcn_wmma_f32_16x16x32_bf16(false, a, false, b, (short)0, c,
                                                 false, false);
}

// ---- WMMA fragment loaders (layouts per CDNA5 ISA §7.12.2) ----
// A (16x32, 16-bit): lane L -> row = L&15, kbase = (L>>4)*8;
//   elems 0..7 -> k = kbase+e ; elems 8..15 -> k = kbase+16+(e-8)
DEV_INLINE v16bf load_a_frag_bf16(const __bf16* __restrict__ A, int ld, int m,
                                  int k0, int lane) {
  const __bf16* p = A + (size_t)(m + (lane & 15)) * ld + (k0 + ((lane >> 4) << 3));
  v16bf r;
#pragma unroll
  for (int e = 0; e < 8; ++e) { r[e] = p[e]; r[e + 8] = p[e + 16]; }
  return r;
}
// B (32x16, 16-bit): lane L -> col = L&15, kbase = (L>>4)*8;
//   elems 0..7 -> k = kbase+e ; elems 8..15 -> k = kbase+16+(e-8)
DEV_INLINE v16bf load_b_frag_bf16(const __bf16* __restrict__ W, int ld, int n,
                                  int k0, int lane) {
  const __bf16* p = W + (size_t)(n + (lane & 15)) * ld + (k0 + ((lane >> 4) << 3));
  v16bf r;
#pragma unroll
  for (int e = 0; e < 8; ++e) { r[e] = p[e]; r[e + 8] = p[e + 16]; }
  return r;
}

// Async global->LDS 16B per lane (CDNA5, tracked by ASYNCcnt).
DEV_INLINE void async_copy_b128(uint32_t lds_off, const void* gaddr) {
  asm volatile("global_load_async_to_lds_b128 %0, %1, off" ::"v"(lds_off),
               "v"(gaddr)
               : "memory");
}
DEV_INLINE uint32_t lds_addr(const void* p) { return (uint32_t)(uintptr_t)p; }
// Async loads complete in order: <=4 outstanding => all older tiles landed.
DEV_INLINE void wait_async_le4() { asm volatile("s_wait_asynccnt 4" ::: "memory"); }
DEV_INLINE void wait_async_0()   { asm volatile("s_wait_asynccnt 0" ::: "memory"); }

// ---- f32 -> bf16 bulk convert (done once; removes cvt from all GEMM loops) ----
__global__ __launch_bounds__(256) void cvt_f32_to_bf16(const float* __restrict__ in,
                                                       __bf16* __restrict__ out,
                                                       int n) {
  const int i = (blockIdx.x * 256 + threadIdx.x) * 4;
  if (i + 3 < n) {
    const float4 v = *(const float4*)(in + i);
    bf16x4 o;
    o.x = (__bf16)v.x; o.y = (__bf16)v.y; o.z = (__bf16)v.z; o.w = (__bf16)v.w;
    *(bf16x4*)(out + i) = o;
  }
}

// ---- Generic bf16 GEMM: C[M,N] = A[M,K] @ W[N,K]^T + bias ----
// Triple-buffered async-LDS staging (DMA issued before waiting on the current
// tile); block 256 thr = 8 waves (4M x 2N), wave tile 32x64, block tile
// 128x128, K step 32.  8 WMMAs per 6 fragment loads.
// OUT_MODE 0: f32 row-major; 1: bf16 q [b,h,s,d]; 2: bf16 k [b,h,s,d] + vT [b,h,d,s]
template <int OUT_MODE>
__global__ __launch_bounds__(256) void gemm_bf16_wmma(
    const __bf16* __restrict__ A, const __bf16* __restrict__ W,
    const float* __restrict__ bias, void* __restrict__ out0,
    void* __restrict__ out1, int M, int N, int K) {
  __shared__ __bf16 Al[3][128 * LDA];  // [m][k] padded rows
  __shared__ __bf16 Wl[3][128 * LDA];  // [n][k] padded rows
  const int lane = threadIdx.x & 31;
  const int wave = threadIdx.x >> 5;
  const int mW = (wave & 3) * 32;   // wave tile origin inside block tile
  const int nW = (wave >> 2) * 64;
  const int mBlk = blockIdx.y * 128;
  const int nBlk = blockIdx.x * 128;

  auto stage = [&](int buf, int k0) {
    const int r = threadIdx.x >> 2;          // 0..63
    const int c = (threadIdx.x & 3) * 16;    // byte within 64B of payload
    const char* ga = (const char*)(A + (size_t)mBlk * K + k0);
    const uint32_t ab = lds_addr(&Al[buf][0]);
    async_copy_b128(ab + r * 80u + c, ga + (size_t)r * (K * 2) + c);
    async_copy_b128(ab + (r + 64) * 80u + c, ga + (size_t)(r + 64) * (K * 2) + c);
    const char* gw = (const char*)(W + (size_t)nBlk * K + k0);
    const uint32_t wb = lds_addr(&Wl[buf][0]);
    async_copy_b128(wb + r * 80u + c, gw + (size_t)r * (K * 2) + c);
    async_copy_b128(wb + (r + 64) * 80u + c, gw + (size_t)(r + 64) * (K * 2) + c);
  };

  v8f acc[2][4] = {};
  const int nIter = K / 32;
  stage(0, 0);
  int cbuf = 0, sbuf = 1;
  for (int it = 0; it < nIter; ++it) {
    if (it + 1 < nIter) {      // issue next DMA first, then wait current tile
      stage(sbuf, (it + 1) * 32);
      wait_async_le4();
    } else {
      wait_async_0();
    }
    __syncthreads();
    v16bf a0 = load_a_frag_bf16(&Al[cbuf][0], LDA, mW, 0, lane);
    v16bf a1 = load_a_frag_bf16(&Al[cbuf][0], LDA, mW + 16, 0, lane);
    v16bf bf[4];
#pragma unroll
    for (int j = 0; j < 4; ++j)
      bf[j] = load_b_frag_bf16(&Wl[cbuf][0], LDA, nW + j * 16, 0, lane);
    __builtin_amdgcn_sched_barrier(0);  // keep ds loads batched ahead of WMMAs
#pragma unroll
    for (int j = 0; j < 4; ++j) {
      acc[0][j] = wmma_bf16(a0, bf[j], acc[0][j]);
      acc[1][j] = wmma_bf16(a1, bf[j], acc[1][j]);
    }
    cbuf = sbuf;
    sbuf = (sbuf == 2) ? 0 : sbuf + 1;
  }
#pragma unroll
  for (int mt = 0; mt < 2; ++mt)
#pragma unroll
    for (int nt = 0; nt < 4; ++nt) {
      const int col = nBlk + nW + nt * 16 + (lane & 15);
      const int r0 = mBlk + mW + mt * 16 + ((lane >> 4) << 3);
      const float bv = bias[col];
#pragma unroll
      for (int v = 0; v < 8; ++v) {
        const int row = r0 + v;
        const float val = acc[mt][nt][v] + bv;
        if constexpr (OUT_MODE == 0) {
          ((float*)out0)[(size_t)row * N + col] = val;
        } else if constexpr (OUT_MODE == 1) {
          const int b = row >> 11, s = row & (SS - 1);
          const int h = col >> 7, d = col & 127;
          ((__bf16*)out0)[(((size_t)(b * HH + h) * SS + s) << 7) + d] = (__bf16)val;
        } else {
          const int b = row >> 11, s = row & (SS - 1);
          const int h = col >> 8, j = col & 255;
          if (j < 128)
            ((__bf16*)out0)[(((size_t)(b * HH + h) * SS + s) << 7) + j] = (__bf16)val;
          else
            ((__bf16*)out1)[((size_t)(b * HH + h) * DVV + (j - 128)) * SS + s] =
                (__bf16)val;
        }
      }
    }
}

// ---- RMSNorm rows of RR=512: f32 in -> bf16 out ----
__global__ __launch_bounds__(256) void rmsnorm_rows_bf16(
    const float* __restrict__ X, const float* __restrict__ w,
    __bf16* __restrict__ Y, int ncols) {
  __shared__ float red[8];
  const float* x = X + (size_t)blockIdx.x * ncols;
  __bf16* y = Y + (size_t)blockIdx.x * ncols;
  float ss = 0.f;
  for (int i = threadIdx.x; i < ncols; i += 256) { float v = x[i]; ss += v * v; }
#pragma unroll
  for (int m = 16; m >= 1; m >>= 1) ss += __shfl_xor(ss, m, 32);
  if ((threadIdx.x & 31) == 0) red[threadIdx.x >> 5] = ss;
  __syncthreads();
  if (threadIdx.x < 8) {
    float t = red[threadIdx.x];
#pragma unroll
    for (int m = 4; m >= 1; m >>= 1) t += __shfl_xor(t, m, 32);
    if (threadIdx.x == 0) red[0] = t;
  }
  __syncthreads();
  const float sc = rsqrtf(red[0] / (float)ncols + 1e-6f);
  for (int i = threadIdx.x; i < ncols; i += 256) y[i] = (__bf16)(w[i] * (x[i] * sc));
}

// ---- Flash attention (causal), bf16 WMMA, exp2-domain softmax ----
// Q/K: [b,h,s,d]; Vt: [b,h,d,s]; O(bf16): [b*s, h*dv]
// 256 thr (8 waves) / 128 q rows; 32-key K/V tiles triple-buffered via
// async-LDS with early DMA issue; padded LDS rows; sched_barrier-pinned
// load/WMMA groups.
__global__ __launch_bounds__(256) void mla_flash_wmma(
    const __bf16* __restrict__ Qbf, const __bf16* __restrict__ Kbf,
    const __bf16* __restrict__ Vt, __bf16* __restrict__ O) {
  __shared__ __bf16 Kl[3][32 * LDK];    // [key][dim] padded
  __shared__ __bf16 Vl[3][128 * LDA];   // [dim][key] padded
  __shared__ __bf16 Plds[8][16 * LDA];  // per-wave P tile, padded
  const int lane = threadIdx.x & 31;
  const int wave = threadIdx.x >> 5;
  const int b = blockIdx.y >> 4;
  const int h = blockIdx.y & 15;
  const int qBlock = blockIdx.x * 128;
  const int qBase = qBlock + wave * 16;

  const __bf16* Q = Qbf + (size_t)(b * HH + h) * SS * DHH;
  const __bf16* Kp = Kbf + (size_t)(b * HH + h) * SS * DHH;
  const __bf16* V = Vt + (size_t)(b * HH + h) * DVV * SS;

  v16bf qf[4];  // Q register-resident for whole kernel
#pragma unroll
  for (int kk = 0; kk < 4; ++kk) qf[kk] = load_a_frag_bf16(Q, DHH, qBase, kk * 32, lane);

  v8f acc[8] = {};
  float mrun[8], lrun[8];
#pragma unroll
  for (int v = 0; v < 8; ++v) { mrun[v] = -3.0e38f; lrun[v] = 0.f; }

  auto stage = [&](int buf, int kb0) {
    // K tile: 32 rows x 256B (global contiguous) -> padded 272B rows
    const char* gk = (const char*)Kp + (size_t)kb0 * 256;
    const uint32_t kb = lds_addr(&Kl[buf][0]);
    {
      const int r = threadIdx.x >> 4, c = (threadIdx.x & 15) * 16;
      async_copy_b128(kb + r * 272u + c, gk + (size_t)r * 256 + c);
      async_copy_b128(kb + (r + 16) * 272u + c, gk + (size_t)(r + 16) * 256 + c);
    }
    // V tile: 128 rows x 64B (global stride S*2) -> padded 80B rows
    const char* gv = (const char*)V + (size_t)kb0 * 2;
    const uint32_t vb = lds_addr(&Vl[buf][0]);
    {
      const int r = threadIdx.x >> 2, c = (threadIdx.x & 3) * 16;
      async_copy_b128(vb + r * 80u + c, gv + (size_t)r * (SS * 2) + c);
      async_copy_b128(vb + (r + 64) * 80u + c, gv + (size_t)(r + 64) * (SS * 2) + c);
    }
  };

  const int nIter = (qBlock + 128) / 32;  // causal: keys 0 .. qBlock+127
  stage(0, 0);
  int cbuf = 0, sbuf = 1;
  for (int it = 0; it < nIter; ++it) {
    const int kb0 = it * 32;
    if (it + 1 < nIter) {      // issue next DMA first, then wait current tile
      stage(sbuf, kb0 + 32);
      wait_async_le4();
    } else {
      wait_async_0();
    }
    __syncthreads();

    // All K fragments batched, then all QK^T WMMAs (partial dscnt waits)
    v16bf kf[8];
#pragma unroll
    for (int kk = 0; kk < 4; ++kk) {
      kf[kk * 2] = load_b_frag_bf16(&Kl[cbuf][0], LDK, 0, kk * 32, lane);
      kf[kk * 2 + 1] = load_b_frag_bf16(&Kl[cbuf][0], LDK, 16, kk * 32, lane);
    }
    __builtin_amdgcn_sched_barrier(0);
    v8f s0 = {}, s1 = {};
#pragma unroll
    for (int kk = 0; kk < 4; ++kk) {
      s0 = wmma_bf16(qf[kk], kf[kk * 2], s0);
      s1 = wmma_bf16(qf[kk], kf[kk * 2 + 1], s1);
    }
    // V fragments issued here: fills WMMA->VALU hazard window, latency hidden
    // under the softmax below.
    v16bf vbf[8];
#pragma unroll
    for (int nt = 0; nt < 8; ++nt)
      vbf[nt] = load_b_frag_bf16(&Vl[cbuf][0], LDA, nt * 16, 0, lane);
    __builtin_amdgcn_sched_barrier(0);

    const int key0 = kb0 + (lane & 15);
    const int qr0 = qBase + ((lane >> 4) << 3);
    float rmax[8];
#pragma unroll
    for (int v = 0; v < 8; ++v) {
      const int qr = qr0 + v;
      float x0 = s0[v] * SCALE_LOG2;  // exp2 domain
      float x1 = s1[v] * SCALE_LOG2;
      if (key0 > qr) x0 = -3.0e38f;
      if (key0 + 16 > qr) x1 = -3.0e38f;
      s0[v] = x0; s1[v] = x1;
      float t = fmaxf(x0, x1);
#pragma unroll
      for (int msk = 8; msk >= 1; msk >>= 1) t = fmaxf(t, __shfl_xor(t, msk, 32));
      rmax[v] = t;  // row max, replicated across the 16-lane half
    }
    float alpha[8];
#pragma unroll
    for (int v = 0; v < 8; ++v) {
      const float mnew = fmaxf(mrun[v], rmax[v]);
      alpha[v] = __builtin_amdgcn_exp2f(mrun[v] - mnew);
      mrun[v] = mnew;
      const float p0 = __builtin_amdgcn_exp2f(s0[v] - mnew);
      const float p1 = __builtin_amdgcn_exp2f(s1[v] - mnew);
      s0[v] = p0; s1[v] = p1;
      float t = p0 + p1;
#pragma unroll
      for (int msk = 8; msk >= 1; msk >>= 1) t += __shfl_xor(t, msk, 32);
      lrun[v] = lrun[v] * alpha[v] + t;
    }

    // C-layout P -> LDS (wave-private, padded) -> A-fragment layout
    __bf16* P = &Plds[wave][0];
#pragma unroll
    for (int v = 0; v < 8; ++v) {
      const int r = ((lane >> 4) << 3) + v;
      P[r * LDA + (lane & 15)] = (__bf16)s0[v];
      P[r * LDA + (lane & 15) + 16] = (__bf16)s1[v];
    }
    asm volatile("s_wait_dscnt 0" ::: "memory");  // wave-local LDS RAW fence
    v16bf pa;
    {
      const int r = lane & 15;
      const int kb = ((lane >> 4) << 3);
      const __bf16* pr = &P[r * LDA];
#pragma unroll
      for (int e = 0; e < 8; ++e) { pa[e] = pr[kb + e]; pa[e + 8] = pr[kb + 16 + e]; }
    }
    // alpha-rescale of O accumulator here hides the P-tile ds round-trip.
#pragma unroll
    for (int nt = 0; nt < 8; ++nt)
#pragma unroll
      for (int v = 0; v < 8; ++v) acc[nt][v] *= alpha[v];
#pragma unroll
    for (int nt = 0; nt < 8; ++nt) acc[nt] = wmma_bf16(pa, vbf[nt], acc[nt]);

    cbuf = sbuf;
    sbuf = (sbuf == 2) ? 0 : sbuf + 1;
  }
  float inv[8];
#pragma unroll
  for (int v = 0; v < 8; ++v) inv[v] = __builtin_amdgcn_rcpf(lrun[v]);
#pragma unroll
  for (int nt = 0; nt < 8; ++nt) {
    const int col = h * DVV + nt * 16 + (lane & 15);
    const int r0 = qBase + ((lane >> 4) << 3);
#pragma unroll
    for (int v = 0; v < 8; ++v)
      O[(size_t)(b * SS + r0 + v) * (HH * DVV) + col] = (__bf16)(acc[nt][v] * inv[v]);
  }
}

extern "C" void kernel_launch(void* const* d_in, const int* in_sizes, int n_in,
                              void* d_out, int out_size, void* d_ws, size_t ws_size,
                              hipStream_t stream) {
  const float* x = (const float*)d_in[0];
  // d_in[1] = mask (causal; implemented analytically), d_in[11] = start_pos (0)
  const float* wq_w = (const float*)d_in[2];
  const float* wq_b = (const float*)d_in[3];
  const float* wkv_a_w = (const float*)d_in[4];
  const float* wkv_a_b = (const float*)d_in[5];
  const float* kv_norm_w = (const float*)d_in[6];
  const float* wkv_b_w = (const float*)d_in[7];
  const float* wkv_b_b = (const float*)d_in[8];
  const float* wo_w = (const float*)d_in[9];
  const float* wo_b = (const float*)d_in[10];
  float* out = (float*)d_out;
  (void)in_sizes; (void)n_in; (void)out_size; (void)ws_size;

  char* ws = (char*)d_ws;
  size_t off = 0;
  auto carve = [&](size_t bytes) -> void* {
    off = (off + 255) & ~(size_t)255;
    void* p = ws + off;
    off += bytes;
    return p;
  };
  const size_t T = (size_t)BB * SS;  // 4096 tokens
  __bf16* x_bf = (__bf16*)carve(T * 2048 * 2);
  __bf16* wq_bf = (__bf16*)carve((size_t)2048 * 2048 * 2);
  __bf16* wkva_bf = (__bf16*)carve((size_t)512 * 2048 * 2);
  __bf16* wkvb_bf = (__bf16*)carve((size_t)4096 * 512 * 2);
  __bf16* wo_bf = (__bf16*)carve((size_t)2048 * 2048 * 2);
  __bf16* q_bf = (__bf16*)carve(T * (HH * DHH) * 2);
  float* kv_c = (float*)carve(T * RR * 4);
  __bf16* kvc_bf = (__bf16*)carve(T * RR * 2);
  __bf16* k_bf = (__bf16*)carve((size_t)BB * HH * SS * DHH * 2);
  __bf16* vT_bf = (__bf16*)carve((size_t)BB * HH * DVV * SS * 2);
  __bf16* attn_bf = (__bf16*)carve(T * (HH * DVV) * 2);

  dim3 blk(256);
  // 0. one-time f32 -> bf16 conversions (activations + weights)
  cvt_f32_to_bf16<<<dim3(8192), blk, 0, stream>>>(x, x_bf, 8388608);
  cvt_f32_to_bf16<<<dim3(4096), blk, 0, stream>>>(wq_w, wq_bf, 4194304);
  cvt_f32_to_bf16<<<dim3(1024), blk, 0, stream>>>(wkv_a_w, wkva_bf, 1048576);
  cvt_f32_to_bf16<<<dim3(2048), blk, 0, stream>>>(wkv_b_w, wkvb_bf, 2097152);
  cvt_f32_to_bf16<<<dim3(4096), blk, 0, stream>>>(wo_w, wo_bf, 4194304);
  // 1. q = x @ wq^T  -> bf16 [b,h,s,d]
  gemm_bf16_wmma<1><<<dim3(2048 / 128, 4096 / 128), blk, 0, stream>>>(
      x_bf, wq_bf, wq_b, q_bf, nullptr, 4096, 2048, 2048);
  // 2. kv_c = x @ wkv_a^T (f32)
  gemm_bf16_wmma<0><<<dim3(512 / 128, 4096 / 128), blk, 0, stream>>>(
      x_bf, wkva_bf, wkv_a_b, kv_c, nullptr, 4096, 512, 2048);
  // 3. rmsnorm rows -> bf16
  rmsnorm_rows_bf16<<<dim3(4096), blk, 0, stream>>>(kv_c, kv_norm_w, kvc_bf, 512);
  // 4. kv = kv_c @ wkv_b^T -> k_bf [b,h,s,d], vT_bf [b,h,d,s]
  gemm_bf16_wmma<2><<<dim3(4096 / 128, 4096 / 128), blk, 0, stream>>>(
      kvc_bf, wkvb_bf, wkv_b_b, k_bf, vT_bf, 4096, 4096, 512);
  // 5. causal flash attention (async-LDS staged K/V) -> bf16
  mla_flash_wmma<<<dim3(SS / 128, BB * HH), blk, 0, stream>>>(q_bf, k_bf, vT_bf,
                                                              attn_bf);
  // 6. out = attn @ wo^T + bias (f32)
  gemm_bf16_wmma<0><<<dim3(2048 / 128, 4096 / 128), blk, 0, stream>>>(
      attn_bf, wo_bf, wo_b, out, nullptr, 4096, 2048, 2048);
}
